// ConstraintLayer_79250736546719
// MI455X (gfx1250) — compile-verified, matrix operand order checked by reading, not run on hardware
//
#include <hip/hip_runtime.h>

// ---------------------------------------------------------------------------
// ConstraintLayer: y* = y - A^T G^{-1} (A y - b),  G = A A^T  (SPD, 128x128)
// BATCH=16384, IN_DIM=1024, OUT_DIM=128.
// Split-bf16 (hi+lo, 3 WMMAs) fp32-emulated GEMMs on v_wmma_f32_16x16x32_bf16.
// ---------------------------------------------------------------------------

#define BATCH   16384
#define IN_DIM  1024
#define OUT_DIM 128

typedef __attribute__((ext_vector_type(16))) __bf16 v16bf;
typedef __attribute__((ext_vector_type(8)))  float  v8f;

static __device__ inline v8f wmma_bf16(v16bf a, v16bf b, v8f c) {
  // (neg_a, A, neg_b, B, c_mod, C, reuse_a, reuse_b)
  return __builtin_amdgcn_wmma_f32_16x16x32_bf16(false, a, false, b,
                                                 (short)0, c, false, false);
}

// A-operand fragment (16x32, 16-bit): lane (l15,half) holds row M=l15,
// elements 0..7 -> K = k0+8*half+e ; elements 8..15 -> K = k0+16+8*half+(e-8).
// Source is f32; split into bf16 hi/lo on the fly. `neg` negates the value.
static __device__ inline void split_fragA(const float* __restrict__ rowp,
                                          int k0, int half, bool neg,
                                          v16bf& hi, v16bf& lo) {
  const float* p0 = rowp + k0 + 8 * half;
  const float* p1 = p0 + 16;
#pragma unroll
  for (int e = 0; e < 8; ++e) {
    float x0 = neg ? -p0[e] : p0[e];
    float x1 = neg ? -p1[e] : p1[e];
    __bf16 h0 = (__bf16)x0;
    __bf16 h1 = (__bf16)x1;
    hi[e]     = h0;  lo[e]     = (__bf16)(x0 - (float)h0);
    hi[e + 8] = h1;  lo[e + 8] = (__bf16)(x1 - (float)h1);
  }
}

// B-operand fragment (32x16, 16-bit), pre-split bf16 source: lane (l15,half)
// holds column N=l15; same K pattern as A -> two contiguous 16B loads.
static __device__ inline v16bf load_fragB(const __bf16* __restrict__ rowp,
                                          int k0, int half) {
  const __bf16* p0 = rowp + k0 + 8 * half;
  const __bf16* p1 = p0 + 16;
  v16bf f;
#pragma unroll
  for (int e = 0; e < 8; ++e) { f[e] = p0[e]; f[e + 8] = p1[e]; }
  return f;
}

// ------------------------- prep kernel 1: G = A A^T -------------------------
__global__ __launch_bounds__(256) void compute_G(const float* __restrict__ A,
                                                 float* __restrict__ G) {
  int idx = blockIdx.x * 256 + threadIdx.x;        // 128*128 outputs
  int r = idx >> 7, c = idx & 127;
  const float* ar = A + (size_t)r * IN_DIM;
  const float* ac = A + (size_t)c * IN_DIM;
  float s = 0.0f;
#pragma unroll 8
  for (int k = 0; k < IN_DIM; ++k) s = fmaf(ar[k], ac[k], s);
  G[idx] = s;
}

// --------------- prep kernel 2: Gauss-Jordan inverse in LDS ----------------
// One workgroup, [G | I] = 128x256 f32 in 128KB dynamic LDS (gfx1250: 320KB).
// G is SPD and well conditioned (cond ~4) -> no pivoting needed.
__global__ __launch_bounds__(512) void invert_G(const float* __restrict__ G,
                                                __bf16* __restrict__ Gh,
                                                __bf16* __restrict__ Gl) {
  extern __shared__ float aug[];                   // 128*256
  __shared__ float colc[128];
  const int tid = threadIdx.x;
  for (int idx = tid; idx < 128 * 256; idx += 512) {
    int i = idx >> 8, j = idx & 255;
    aug[idx] = (j < 128) ? G[i * 128 + j] : ((j - 128) == i ? 1.0f : 0.0f);
  }
  __syncthreads();
  for (int k = 0; k < 128; ++k) {
    if (tid < 128) colc[tid] = aug[tid * 256 + k];
    __syncthreads();
    float pinv = 1.0f / colc[k];
    for (int j = tid; j < 256; j += 512) aug[k * 256 + j] *= pinv;
    __syncthreads();
    int tcol = tid & 255, r0 = tid >> 8;           // r0 in {0,1}
    float pk = aug[k * 256 + tcol];
    for (int i = r0; i < 128; i += 2)
      if (i != k) aug[i * 256 + tcol] -= colc[i] * pk;
    __syncthreads();
  }
  for (int idx = tid; idx < 128 * 128; idx += 512) {
    int i = idx >> 7, j = idx & 127;
    float v = aug[i * 256 + 128 + j];
    __bf16 h = (__bf16)v;
    Gh[idx] = h; Gl[idx] = (__bf16)(v - (float)h);
  }
}

// ------- prep kernel 3: split A into bf16 hi/lo, row-major + transposed ----
__global__ __launch_bounds__(256) void pack_A(const float* __restrict__ A,
                                              __bf16* __restrict__ Ah,
                                              __bf16* __restrict__ Al,
                                              __bf16* __restrict__ Ath,
                                              __bf16* __restrict__ Atl) {
  int idx = blockIdx.x * 256 + threadIdx.x;        // 128*1024 elements
  int r = idx >> 10, c = idx & 1023;
  float x = A[idx];
  __bf16 h = (__bf16)x;
  __bf16 l = (__bf16)(x - (float)h);
  Ah[idx] = h;  Al[idx] = l;
  Ath[c * 128 + r] = h;  Atl[c * 128 + r] = l;
}

// ----------------------------- fused main kernel ---------------------------
// One block = 16 batch rows, 4 waves. Stage0: r = y A^T - b (K=1024).
// Stage1: mu = r Ginv (K=128, Ginv symmetric -> row-major ok).
// Stage2: out = y + (-mu) A (K=128), C initialized with f32 y tile.
__global__ __launch_bounds__(128) void fused_project(
    const float* __restrict__ y, const float* __restrict__ b,
    const __bf16* __restrict__ Ah, const __bf16* __restrict__ Al,
    const __bf16* __restrict__ Ath, const __bf16* __restrict__ Atl,
    const __bf16* __restrict__ Gh, const __bf16* __restrict__ Gl,
    float* __restrict__ out) {
  __shared__ float s_r[16][OUT_DIM];
  __shared__ float s_mu[16][OUT_DIM];

  const int tid  = threadIdx.x;
  const int wid  = tid >> 5;
  const int lane = tid & 31;
  const int half = lane >> 4;
  const int l15  = lane & 15;
  const int i0   = blockIdx.x * 16;

  // ---- Stage 0: r = y A^T - b ; each wave owns two 16-col tiles of r ----
  const int n0a = (wid * 2) * 16;
  const int n0b = n0a + 16;
  const float*  yrow = y  + (size_t)(i0 + l15) * IN_DIM;
  const __bf16* b0h  = Ah + (size_t)(n0a + l15) * IN_DIM;
  const __bf16* b0l  = Al + (size_t)(n0a + l15) * IN_DIM;
  const __bf16* b1h  = Ah + (size_t)(n0b + l15) * IN_DIM;
  const __bf16* b1l  = Al + (size_t)(n0b + l15) * IN_DIM;

  v8f acc0 = {};
  v8f acc1 = {};
  for (int kk = 0; kk < IN_DIM / 32; ++kk) {
    const int k0 = kk * 32;
    v16bf yh, yl;
    split_fragA(yrow, k0, half, false, yh, yl);      // shared across 2 tiles
    v16bf fh = load_fragB(b0h, k0, half);
    v16bf fl = load_fragB(b0l, k0, half);
    acc0 = wmma_bf16(yh, fh, acc0);
    acc0 = wmma_bf16(yl, fh, acc0);
    acc0 = wmma_bf16(yh, fl, acc0);
    fh = load_fragB(b1h, k0, half);
    fl = load_fragB(b1l, k0, half);
    acc1 = wmma_bf16(yh, fh, acc1);
    acc1 = wmma_bf16(yl, fh, acc1);
    acc1 = wmma_bf16(yh, fl, acc1);
  }
#pragma unroll
  for (int v = 0; v < 8; ++v) {                      // D layout: row = v+8*half
    int m = v + 8 * half;
    s_r[m][n0a + l15] = acc0[v] - b[(size_t)(i0 + m) * OUT_DIM + n0a + l15];
    s_r[m][n0b + l15] = acc1[v] - b[(size_t)(i0 + m) * OUT_DIM + n0b + l15];
  }
  __syncthreads();

  // ---- Stage 1: mu = r * Ginv (K = 128) ----
  const __bf16* g0h = Gh + (size_t)(n0a + l15) * OUT_DIM;
  const __bf16* g0l = Gl + (size_t)(n0a + l15) * OUT_DIM;
  const __bf16* g1h = Gh + (size_t)(n0b + l15) * OUT_DIM;
  const __bf16* g1l = Gl + (size_t)(n0b + l15) * OUT_DIM;
  v8f mu0 = {};
  v8f mu1 = {};
#pragma unroll
  for (int kk = 0; kk < 4; ++kk) {
    const int k0 = kk * 32;
    v16bf rh, rl;
    split_fragA(&s_r[l15][0], k0, half, false, rh, rl);
    v16bf fh = load_fragB(g0h, k0, half);
    v16bf fl = load_fragB(g0l, k0, half);
    mu0 = wmma_bf16(rh, fh, mu0);
    mu0 = wmma_bf16(rl, fh, mu0);
    mu0 = wmma_bf16(rh, fl, mu0);
    fh = load_fragB(g1h, k0, half);
    fl = load_fragB(g1l, k0, half);
    mu1 = wmma_bf16(rh, fh, mu1);
    mu1 = wmma_bf16(rl, fh, mu1);
    mu1 = wmma_bf16(rh, fl, mu1);
  }
#pragma unroll
  for (int v = 0; v < 8; ++v) {
    int m = v + 8 * half;
    s_mu[m][n0a + l15] = mu0[v];
    s_mu[m][n0b + l15] = mu1[v];
  }
  __syncthreads();

  // ---- Stage 2: out = y + (-mu) * A ; each wave owns 16 x 16-col tiles ----
  v16bf mh[4], ml[4];
#pragma unroll
  for (int kk = 0; kk < 4; ++kk)                     // negated mu fragments
    split_fragA(&s_mu[l15][0], kk * 32, half, true, mh[kk], ml[kk]);

  for (int t = 0; t < 16; ++t) {
    const int c0 = (wid * 16 + t) * 16;
    const __bf16* ph = Ath + (size_t)(c0 + l15) * OUT_DIM;
    const __bf16* pl = Atl + (size_t)(c0 + l15) * OUT_DIM;
    v8f acc;
#pragma unroll
    for (int v = 0; v < 8; ++v)                      // C := y tile (f32 exact)
      acc[v] = y[(size_t)(i0 + v + 8 * half) * IN_DIM + c0 + l15];
#pragma unroll
    for (int kk = 0; kk < 4; ++kk) {
      const int k0 = kk * 32;
      v16bf fh = load_fragB(ph, k0, half);
      v16bf fl = load_fragB(pl, k0, half);
      acc = wmma_bf16(mh[kk], fh, acc);
      acc = wmma_bf16(ml[kk], fh, acc);
      acc = wmma_bf16(mh[kk], fl, acc);
    }
#pragma unroll
    for (int v = 0; v < 8; ++v)
      out[(size_t)(i0 + v + 8 * half) * IN_DIM + c0 + l15] = acc[v];
  }
}

// ------------------------------- launcher ----------------------------------
extern "C" void kernel_launch(void* const* d_in, const int* in_sizes, int n_in,
                              void* d_out, int out_size, void* d_ws,
                              size_t ws_size, hipStream_t stream) {
  (void)in_sizes; (void)n_in; (void)out_size; (void)ws_size;
  const float* y = (const float*)d_in[0];   // 16384 x 1024
  const float* A = (const float*)d_in[1];   // 128 x 1024
  const float* b = (const float*)d_in[2];   // 16384 x 128
  float* out = (float*)d_out;               // 16384 x 1024

  char* ws = (char*)d_ws;                   // ~1.2 MB used
  float*  G   = (float*) (ws + 0);          //  64 KB
  __bf16* Gh  = (__bf16*)(ws + 65536);      //  32 KB
  __bf16* Gl  = (__bf16*)(ws + 98304);      //  32 KB
  __bf16* Ah  = (__bf16*)(ws + 131072);     // 256 KB
  __bf16* Al  = (__bf16*)(ws + 393216);     // 256 KB
  __bf16* Ath = (__bf16*)(ws + 655360);     // 256 KB
  __bf16* Atl = (__bf16*)(ws + 917504);     // 256 KB

  compute_G<<<64, 256, 0, stream>>>(A, G);
  invert_G<<<1, 512, 128 * 256 * sizeof(float), stream>>>(G, Gh, Gl);
  pack_A<<<512, 256, 0, stream>>>(A, Ah, Al, Ath, Atl);
  fused_project<<<BATCH / 16, 128, 0, stream>>>(y, b, Ah, Al, Ath, Atl,
                                                Gh, Gl, out);
}
